// RD_Encoder_1176821039847
// MI455X (gfx1250) — matrix operand order repaired and checked by reading, not compiled, not run on hardware
//
#include <hip/hip_runtime.h>
#include <math.h>
#include <stdint.h>

typedef __attribute__((ext_vector_type(16))) _Float16 v16h;
typedef __attribute__((ext_vector_type(8)))  _Float16 v8h;
typedef __attribute__((ext_vector_type(8)))  float    v8f;

enum { EP_NONE = 0, EP_GELU = 1, EP_RESID_GAMMA = 2 };

__device__ __forceinline__ float gelu_exact(float x) {
    return 0.5f * x * (1.0f + erff(x * 0.70710678118654752f));
}
__device__ __forceinline__ float sigmf(float x) {
    return 1.0f / (1.0f + expf(-x));
}

__device__ __forceinline__ void wait_asynccnt0() {
#if __has_builtin(__builtin_amdgcn_s_wait_asynccnt)
    __builtin_amdgcn_s_wait_asynccnt(0);
#else
    asm volatile("s_wait_asynccnt 0" ::: "memory");
#endif
}

// ---------------------------------------------------------------------------
// WMMA GEMM: out[M,N] = epilogue(A[M,K] * W16[N,K]^T + bias)
// A row-major fp32 (ld=K), W16 row-major f16 (ld=K).  K%32==0.
// Block: 256 thr = 8 waves; wave -> 16 rows x (NT*16) cols; block -> 128 rows.
// The W tile (NT*16 cols x 32 k, f16) is DMA'd into LDS with
// GLOBAL_LOAD_ASYNC_TO_LDS_B128 (ASYNCcnt) each K-step: one b128 per thread,
// no VGPR round-trip, 8x reuse across the block's waves. B fragments are
// single 32-byte aligned LDS reads; all NT are preloaded before the WMMA chain.
// ---------------------------------------------------------------------------
template <int EP, int NT>
__global__ __launch_bounds__(256) void gemm_wmma_kernel(
    const float* __restrict__ A, const _Float16* __restrict__ W,
    const float* __restrict__ bias, float* __restrict__ out,
    int M, int N, int K, int colOfs,
    const float* __restrict__ gamma, const float* __restrict__ resid)
{
    __shared__ __attribute__((aligned(64))) _Float16 lw[64 * 32];  // [col][k]

    const int lane    = threadIdx.x & 31;
    const int wave    = threadIdx.x >> 5;
    const int rowBase = blockIdx.y * 128 + wave * 16;
    const int colBase = colOfs + blockIdx.x * 64;
    const int lrow    = lane & 15;
    const int hi      = lane >> 4;

    v8f acc[NT] = {};

    int arow = rowBase + lrow;
    if (arow >= M) arow = M - 1;  // clamp so all threads reach barriers
    const float* Arow = A + (size_t)arow * K;

    // staging map: 4 threads per column, 8 consecutive f16 (16 bytes) each
    const int  scol   = threadIdx.x >> 2;        // 0..63
    const int  sk     = (threadIdx.x & 3) * 8;   // 0,8,16,24 (halves)
    const bool stager = (scol < NT * 16);        // wave-uniform (64-thr aligned)
    const uint32_t ldsAddr = (uint32_t)(size_t)(&lw[scol * 32 + sk]);
    const _Float16* Wst = W + (size_t)(colBase + (stager ? scol : 0)) * K + sk;

    for (int kb = 0; kb < K; kb += 32) {
        __syncthreads();  // previous iteration's LDS reads complete
        if (stager) {
            uint64_t ga = (uint64_t)(size_t)(Wst + kb);
            asm volatile("global_load_async_to_lds_b128 %0, %1, off"
                         :: "v"(ldsAddr), "v"(ga) : "memory");
        }
        wait_asynccnt0();
        __syncthreads();

        if (kb + 32 < K) {
            __builtin_prefetch(Arow + kb + 32, 0, 0);  // global_prefetch_b8
        }
        // A fragment (16x32 f16): lanes<16 K = {0..15}+kb, lanes>=16 +16
        v16h afrag;
        {
            const float* p = Arow + kb + (hi ? 16 : 0);
            #pragma unroll
            for (int j = 0; j < 8; ++j) {
                int k = 2 * j;
                float2 tv = *(const float2*)(p + k);
                afrag[2 * j]     = (_Float16)tv.x;
                afrag[2 * j + 1] = (_Float16)tv.y;
            }
        }
        // Preload all B fragments, then issue the WMMA chain back-to-back.
        v16h bfr[NT];
        #pragma unroll
        for (int t = 0; t < NT; ++t)
            bfr[t] = *(const v16h*)(&lw[(16 * t + lrow) * 32 + hi * 16]);
        #pragma unroll
        for (int t = 0; t < NT; ++t)
            acc[t] = __builtin_amdgcn_wmma_f32_16x16x32_f16(
                false, afrag, false, bfr[t], (short)0, acc[t], false, false);
    }

    // Epilogue: VGPR v -> row rowBase+v (+8 for hi half), col = colBase+16t+lrow
    const int mofs = hi ? 8 : 0;
    #pragma unroll
    for (int t = 0; t < NT; ++t) {
        int   n  = colBase + 16 * t + lrow;
        float bv = bias ? bias[n] : 0.0f;
        float gv = (EP == EP_RESID_GAMMA) ? gamma[n] : 1.0f;
        #pragma unroll
        for (int v = 0; v < 8; ++v) {
            int r = rowBase + v + mofs;
            if (r < M) {
                float val = acc[t][v] + bv;
                if (EP == EP_GELU)             val = gelu_exact(val);
                else if (EP == EP_RESID_GAMMA) val = resid[(size_t)r * N + n] + val * gv;
                out[(size_t)r * N + n] = val;
            }
        }
    }
}

// ---------------------------------------------------------------------------
// fp32 -> f16 weight conversion (run once per call, per GEMM weight)
// ---------------------------------------------------------------------------
__global__ void f32_to_f16_kernel(const float* __restrict__ in,
                                  _Float16* __restrict__ out, long n)
{
    long i = (long)blockIdx.x * blockDim.x + threadIdx.x;
    if (i < n) out[i] = (_Float16)in[i];
}

// ---------------------------------------------------------------------------
// LayerNorm over last dim (one wave32 per row)
// ---------------------------------------------------------------------------
__global__ __launch_bounds__(256) void layernorm_kernel(
    const float* __restrict__ in, float* __restrict__ out,
    const float* __restrict__ g, const float* __restrict__ b,
    int rows, int C, float eps)
{
    int wave = threadIdx.x >> 5, lane = threadIdx.x & 31;
    int row = blockIdx.x * 8 + wave;
    if (row >= rows) return;
    const float* p = in + (size_t)row * C;
    float s = 0.f, s2 = 0.f;
    for (int c = lane; c < C; c += 32) { float v = p[c]; s += v; s2 += v * v; }
    #pragma unroll
    for (int o = 16; o > 0; o >>= 1) {
        s  += __shfl_xor(s, o, 32);
        s2 += __shfl_xor(s2, o, 32);
    }
    float mean = s / (float)C;
    float var  = s2 / (float)C - mean * mean;
    float rstd = rsqrtf(var + eps);
    float* q = out + (size_t)row * C;
    for (int c = lane; c < C; c += 32) q[c] = (p[c] - mean) * rstd * g[c] + b[c];
}

// ---------------------------------------------------------------------------
// Stem: 4x4 stride-4 conv, 1 in-channel -> 96, output NHWC (256,16,16,96)
// ---------------------------------------------------------------------------
__global__ void stem_kernel(const float* __restrict__ x, const float* __restrict__ w,
                            const float* __restrict__ b, float* __restrict__ out)
{
    int idx = blockIdx.x * blockDim.x + threadIdx.x;
    const int total = 256 * 16 * 16 * 96;
    if (idx >= total) return;
    int oc = idx % 96; int t = idx / 96;
    int ox = t % 16; t /= 16;
    int oy = t % 16; int n = t / 16;
    const float* xp = x + ((size_t)n * 64 + oy * 4) * 64 + ox * 4;
    const float* wp = w + oc * 16;
    float acc = b[oc];
    #pragma unroll
    for (int ky = 0; ky < 4; ++ky)
        #pragma unroll
        for (int kx = 0; kx < 4; ++kx)
            acc += xp[ky * 64 + kx] * wp[ky * 4 + kx];
    out[idx] = acc;
}

// ---------------------------------------------------------------------------
// Depthwise 7x7 conv, pad 3, NHWC
// ---------------------------------------------------------------------------
__global__ void dwconv7_kernel(const float* __restrict__ in, const float* __restrict__ w,
                               const float* __restrict__ b, float* __restrict__ out,
                               int NI, int S, int C)
{
    long idx = (long)blockIdx.x * blockDim.x + threadIdx.x;
    long total = (long)NI * S * S * C;
    if (idx >= total) return;
    int c = (int)(idx % C); long t = idx / C;
    int x = (int)(t % S); t /= S;
    int y = (int)(t % S); int n = (int)(t / S);
    float acc = b[c];
    const float* wp = w + (size_t)c * 49;
    for (int ky = 0; ky < 7; ++ky) {
        int iy = y + ky - 3; if (iy < 0 || iy >= S) continue;
        for (int kx = 0; kx < 7; ++kx) {
            int ix = x + kx - 3; if (ix < 0 || ix >= S) continue;
            acc += in[(((size_t)n * S + iy) * S + ix) * C + c] * wp[ky * 7 + kx];
        }
    }
    out[idx] = acc;
}

// ---------------------------------------------------------------------------
// Downsample helpers: 2x2 stride-2 conv as repacked-f16-weight GEMM
// ---------------------------------------------------------------------------
__global__ void im2col2x2_kernel(const float* __restrict__ in, float* __restrict__ out,
                                 int NI, int S, int C)
{
    long idx = (long)blockIdx.x * blockDim.x + threadIdx.x;
    int So = S / 2;
    long total = (long)NI * So * So * 4 * C;
    if (idx >= total) return;
    int c = (int)(idx % C); long t = idx / C;
    int kx = (int)(t & 1); t >>= 1;
    int ky = (int)(t & 1); t >>= 1;
    int ox = (int)(t % So); t /= So;
    int oy = (int)(t % So); int n = (int)(t / So);
    out[idx] = in[(((size_t)n * S + (2 * oy + ky)) * S + (2 * ox + kx)) * C + c];
}

__global__ void wrepack2x2_kernel(const float* __restrict__ w, _Float16* __restrict__ out,
                                  int Cout, int Cin)
{
    int idx = blockIdx.x * blockDim.x + threadIdx.x;
    int total = Cout * Cin * 4;
    if (idx >= total) return;
    int c = idx % Cin; int t = idx / Cin;
    int kx = t & 1; t >>= 1;
    int ky = t & 1; t >>= 1;
    int o = t;
    out[idx] = (_Float16)w[(((size_t)o * Cin + c) * 2 + ky) * 2 + kx];
}

// ---------------------------------------------------------------------------
// Global average pool (256,2,2,768) -> (256,768); mask from lengths
// ---------------------------------------------------------------------------
__global__ void avgpool_kernel(const float* __restrict__ in, float* __restrict__ out)
{
    int idx = blockIdx.x * blockDim.x + threadIdx.x;
    if (idx >= 256 * 768) return;
    int c = idx % 768, n = idx / 768;
    const float* p = in + (size_t)n * 4 * 768 + c;
    out[idx] = 0.25f * (p[0] + p[768] + p[2 * 768] + p[3 * 768]);
}

__global__ void mask_kernel(const int* __restrict__ lengths, float* __restrict__ m)
{
    int idx = threadIdx.x;  // 256 = B*T
    int b = idx / 32, t = idx % 32;
    m[idx] = (t < lengths[b]) ? 1.0f : 0.0f;
}

// ---------------------------------------------------------------------------
// Mamba pieces (Hh=1536, S=8, T=32, B=8)
// ---------------------------------------------------------------------------
__global__ void mamba_gateconv_kernel(const float* __restrict__ ip,
                                      const float* __restrict__ cw, const float* __restrict__ cb,
                                      float* __restrict__ gate, float* __restrict__ uconv)
{
    int idx = blockIdx.x * blockDim.x + threadIdx.x;
    const int Hh = 1536, T = 32;
    if (idx >= 256 * Hh) return;
    int h = idx % Hh; int bt = idx / Hh;
    int b = bt / T, t = bt % T;
    gate[idx] = sigmf(ip[(size_t)bt * 2 * Hh + h]);
    float acc = cb[h];
    #pragma unroll
    for (int k = 0; k < 4; ++k) {
        int tt = t - 2 + k;
        if (tt >= 0 && tt < T) {
            float v = ip[((size_t)(b * T + tt)) * 2 * Hh + Hh + h];
            acc += cw[h * 4 + k] * (v * sigmf(v));  // SiLU then depthwise conv1d
        }
    }
    uconv[idx] = acc;
}

__global__ void bu_kernel(const float* __restrict__ uconv, const float* __restrict__ Bm,
                          const float* __restrict__ m, float* __restrict__ Bu)
{
    __shared__ float red[64];
    int bs = blockIdx.x; int s = bs % 8; int bt = bs / 8;
    float acc = 0.f;
    for (int h = threadIdx.x; h < 1536; h += 64)
        acc += uconv[(size_t)bt * 1536 + h] * Bm[h * 8 + s];
    red[threadIdx.x] = acc;
    __syncthreads();
    for (int o = 32; o > 0; o >>= 1) {
        if ((int)threadIdx.x < o) red[threadIdx.x] += red[threadIdx.x + o];
        __syncthreads();
    }
    if (threadIdx.x == 0) Bu[bt * 8 + s] = red[0] * m[bt];
}

__global__ void cumsum_kernel(const float* __restrict__ Bu, float* __restrict__ hs)
{
    int idx = threadIdx.x;  // 64 = B*S
    if (idx >= 64) return;
    int s = idx % 8, b = idx / 8;
    float acc = 0.f;
    for (int t = 0; t < 32; ++t) {
        acc += Bu[(b * 32 + t) * 8 + s];
        hs[(b * 32 + t) * 8 + s] = acc;
    }
}

__global__ void mamba_outh_kernel(const float* __restrict__ hs, const float* __restrict__ Am,
                                  const float* __restrict__ gate, const float* __restrict__ m,
                                  float* __restrict__ oh)
{
    int idx = blockIdx.x * blockDim.x + threadIdx.x;
    const int Hh = 1536;
    if (idx >= 256 * Hh) return;
    int h = idx % Hh; int bt = idx / Hh;
    float acc = 0.f;
    #pragma unroll
    for (int s = 0; s < 8; ++s) acc += hs[bt * 8 + s] * Am[h * 8 + s];
    oh[idx] = gate[idx] * acc * m[bt];
}

// ---------------------------------------------------------------------------
// Host-side GEMM dispatch: full 64-col tiles (NT=4) + 16/32/48-col remainder
// ---------------------------------------------------------------------------
static void launch_gemm(const float* A, const _Float16* W, const float* bias, float* out,
                        int M, int N, int K, int ep,
                        const float* gamma, const float* resid, hipStream_t s)
{
    int gy  = (M + 127) / 128;
    int n64 = N & ~63;
    if (n64) {
        dim3 g(n64 / 64, gy);
        if      (ep == EP_NONE)
            gemm_wmma_kernel<EP_NONE, 4><<<g, 256, 0, s>>>(A, W, bias, out, M, N, K, 0, gamma, resid);
        else if (ep == EP_GELU)
            gemm_wmma_kernel<EP_GELU, 4><<<g, 256, 0, s>>>(A, W, bias, out, M, N, K, 0, gamma, resid);
        else
            gemm_wmma_kernel<EP_RESID_GAMMA, 4><<<g, 256, 0, s>>>(A, W, bias, out, M, N, K, 0, gamma, resid);
    }
    int rem = N - n64;  // multiple of 16 (all N here are)
    if (rem) {
        dim3 g(1, gy);
        int nt = rem / 16;
        if (ep == EP_NONE) {
            if      (nt == 1) gemm_wmma_kernel<EP_NONE, 1><<<g, 256, 0, s>>>(A, W, bias, out, M, N, K, n64, gamma, resid);
            else if (nt == 2) gemm_wmma_kernel<EP_NONE, 2><<<g, 256, 0, s>>>(A, W, bias, out, M, N, K, n64, gamma, resid);
            else              gemm_wmma_kernel<EP_NONE, 3><<<g, 256, 0, s>>>(A, W, bias, out, M, N, K, n64, gamma, resid);
        } else if (ep == EP_GELU) {
            if      (nt == 1) gemm_wmma_kernel<EP_GELU, 1><<<g, 256, 0, s>>>(A, W, bias, out, M, N, K, n64, gamma, resid);
            else if (nt == 2) gemm_wmma_kernel<EP_GELU, 2><<<g, 256, 0, s>>>(A, W, bias, out, M, N, K, n64, gamma, resid);
            else              gemm_wmma_kernel<EP_GELU, 3><<<g, 256, 0, s>>>(A, W, bias, out, M, N, K, n64, gamma, resid);
        } else {
            if      (nt == 1) gemm_wmma_kernel<EP_RESID_GAMMA, 1><<<g, 256, 0, s>>>(A, W, bias, out, M, N, K, n64, gamma, resid);
            else if (nt == 2) gemm_wmma_kernel<EP_RESID_GAMMA, 2><<<g, 256, 0, s>>>(A, W, bias, out, M, N, K, n64, gamma, resid);
            else              gemm_wmma_kernel<EP_RESID_GAMMA, 3><<<g, 256, 0, s>>>(A, W, bias, out, M, N, K, n64, gamma, resid);
        }
    }
}

// ---------------------------------------------------------------------------
// Host orchestration
// ---------------------------------------------------------------------------
extern "C" void kernel_launch(void* const* d_in, const int* in_sizes, int n_in,
                              void* d_out, int out_size, void* d_ws, size_t ws_size,
                              hipStream_t stream)
{
    (void)in_sizes; (void)n_in; (void)out_size; (void)ws_size;
    int pi = 0;
    auto nextp = [&]() { return (const float*)d_in[pi++]; };

    const float* x        = nextp();                 // (8,32,64,64)
    const int*   lengths  = (const int*)d_in[pi++];  // (8,)
    const float* stem_w = nextp(); const float* stem_b  = nextp();
    const float* stem_g = nextp(); const float* stem_be = nextp();

    struct Blk   { const float *dw_w, *dw_b, *ln_g, *ln_b, *w1, *b1, *w2, *b2, *gamma; };
    struct Stage { const float *dg, *dbe, *dw, *db; Blk blk[9]; int depth; };
    Stage st[4];
    const int depths[4] = {3, 3, 9, 3};
    const int dims[4]   = {96, 192, 384, 768};
    for (int i = 0; i < 4; ++i) {
        st[i].depth = depths[i];
        if (i > 0) { st[i].dg = nextp(); st[i].dbe = nextp(); st[i].dw = nextp(); st[i].db = nextp(); }
        else       { st[i].dg = st[i].dbe = st[i].dw = st[i].db = nullptr; }
        for (int d = 0; d < depths[i]; ++d) {
            Blk& b = st[i].blk[d];
            b.dw_w = nextp(); b.dw_b = nextp(); b.ln_g = nextp(); b.ln_b = nextp();
            b.w1 = nextp(); b.b1 = nextp(); b.w2 = nextp(); b.b2 = nextp(); b.gamma = nextp();
        }
    }
    struct Mam { const float *in_w, *in_b, *cw, *cb, *A, *B, *ow, *ob; } mam[2];
    for (int i = 0; i < 2; ++i) {
        mam[i].in_w = nextp(); mam[i].in_b = nextp(); mam[i].cw = nextp(); mam[i].cb = nextp();
        mam[i].A = nextp(); mam[i].B = nextp(); mam[i].ow = nextp(); mam[i].ob = nextp();
    }
    const float* norm_g = nextp(); const float* norm_b = nextp();
    const float* p1_w = nextp();   const float* p1_b = nextp();
    const float* p2_w = nextp();   const float* p2_b = nextp();

    // Workspace carving (byte allocator; deterministic layout every call)
    char* wsb = (char*)d_ws;
    size_t wsOff = 0;
    auto allocB = [&](size_t bytes) -> void* {
        void* p = wsb + wsOff; wsOff = (wsOff + bytes + 255) & ~(size_t)255; return p;
    };
    auto allocF = [&](size_t n) -> float* { return (float*)allocB(n * 4); };
    auto cvtw = [&](const float* src, size_t n) -> const _Float16* {
        _Float16* dst = (_Float16*)allocB(n * 2);
        f32_to_f16_kernel<<<(int)((n + 511) / 512), 512, 0, stream>>>(src, dst, (long)n);
        return dst;
    };

    const size_t ACT = (size_t)65536 * 96;  // max NHWC activation (stage 1)
    float* bufX = allocF(ACT);
    float* bufY = allocF(ACT);
    float* bufZ = allocF(ACT);
    float* bufH = allocF((size_t)65536 * 384);  // max MLP hidden
    _Float16* wrep = (_Float16*)allocB((size_t)768 * 1536 * 2);
    float* seq  = allocF((size_t)256 * 768);
    float* ip   = allocF((size_t)256 * 3072);
    float* gate = allocF((size_t)256 * 1536);
    float* ucv  = allocF((size_t)256 * 1536);
    float* Bu   = allocF(2048);
    float* hs   = allocF(2048);
    float* oh   = allocF((size_t)256 * 1536);
    float* t1   = allocF((size_t)256 * 768);
    float* msk  = allocF(256);
    // (f16 weight arena grows from here via cvtw)

    const int NI = 256;

    // ---- Stem conv + channel LN ----
    {
        int total = NI * 16 * 16 * 96;
        stem_kernel<<<(total + 255) / 256, 256, 0, stream>>>(x, stem_w, stem_b, bufX);
        int rows = NI * 16 * 16;
        layernorm_kernel<<<(rows + 7) / 8, 256, 0, stream>>>(bufX, bufX, stem_g, stem_be, rows, 96, 1e-6f);
    }

    int S = 16;
    float* cur = bufX;
    float* spare = bufZ;
    for (int i = 0; i < 4; ++i) {
        int C = dims[i];
        if (i > 0) {
            int Cin = dims[i - 1];
            int rows = NI * S * S;
            layernorm_kernel<<<(rows + 7) / 8, 256, 0, stream>>>(cur, cur, st[i].dg, st[i].dbe, rows, Cin, 1e-6f);
            int wt = C * Cin * 4;
            wrepack2x2_kernel<<<(wt + 255) / 256, 256, 0, stream>>>(st[i].dw, wrep, C, Cin);
            long tot = (long)NI * (S / 2) * (S / 2) * 4 * Cin;
            im2col2x2_kernel<<<(int)((tot + 255) / 256), 256, 0, stream>>>(cur, bufY, NI, S, Cin);
            S /= 2;
            int M = NI * S * S, K = 4 * Cin;
            launch_gemm(bufY, wrep, st[i].db, spare, M, C, K, EP_NONE, nullptr, nullptr, stream);
            float* t0 = cur; cur = spare; spare = t0;
        }
        int M = NI * S * S;
        for (int d = 0; d < st[i].depth; ++d) {
            const Blk& b = st[i].blk[d];
            const _Float16* w1h = cvtw(b.w1, (size_t)4 * C * C);
            const _Float16* w2h = cvtw(b.w2, (size_t)4 * C * C);
            long tot = (long)M * C;
            dwconv7_kernel<<<(int)((tot + 255) / 256), 256, 0, stream>>>(cur, b.dw_w, b.dw_b, bufY, NI, S, C);
            layernorm_kernel<<<(M + 7) / 8, 256, 0, stream>>>(bufY, bufY, b.ln_g, b.ln_b, M, C, 1e-6f);
            launch_gemm(bufY, w1h, b.b1, bufH, M, 4 * C, C, EP_GELU, nullptr, nullptr, stream);
            launch_gemm(bufH, w2h, b.b2, spare, M, C, 4 * C, EP_RESID_GAMMA, b.gamma, cur, stream);
            float* t0 = cur; cur = spare; spare = t0;
        }
    }

    // ---- Pool + mask ----
    avgpool_kernel<<<(256 * 768 + 255) / 256, 256, 0, stream>>>(cur, seq);
    mask_kernel<<<1, 256, 0, stream>>>(lengths, msk);

    // ---- Two Mamba layers ----
    for (int l = 0; l < 2; ++l) {
        const Mam& mm = mam[l];
        const _Float16* inwh = cvtw(mm.in_w, (size_t)3072 * 768);
        const _Float16* owh  = cvtw(mm.ow,  (size_t)768 * 1536);
        launch_gemm(seq, inwh, mm.in_b, ip, 256, 3072, 768, EP_NONE, nullptr, nullptr, stream);
        int tot = 256 * 1536;
        mamba_gateconv_kernel<<<(tot + 255) / 256, 256, 0, stream>>>(ip, mm.cw, mm.cb, gate, ucv);
        bu_kernel<<<256 * 8, 64, 0, stream>>>(ucv, mm.B, msk, Bu);
        cumsum_kernel<<<1, 64, 0, stream>>>(Bu, hs);
        mamba_outh_kernel<<<(tot + 255) / 256, 256, 0, stream>>>(hs, mm.A, gate, msk, oh);
        launch_gemm(oh, owh, mm.ob, seq, 256, 768, 1536, EP_NONE, nullptr, nullptr, stream);
    }

    // ---- Head ----
    const _Float16* p1h = cvtw(p1_w, (size_t)768 * 768);
    const _Float16* p2h = cvtw(p2_w, (size_t)128 * 768);
    layernorm_kernel<<<(256 + 7) / 8, 256, 0, stream>>>(seq, seq, norm_g, norm_b, 256, 768, 1e-5f);
    launch_gemm(seq, p1h, p1_b, t1, 256, 768, 768, EP_GELU, nullptr, nullptr, stream);
    launch_gemm(t1, p2h, p2_b, (float*)d_out, 256, 128, 768, EP_NONE, nullptr, nullptr, stream);
}
